// WindowAttention_76785425318403
// MI455X (gfx1250) — compile-verified, matrix operand order checked by reading, not run on hardware
//
#include <hip/hip_runtime.h>

// ---- CDNA5 (gfx1250) wave32 WMMA window-attention ----
// Precision: bf16 inputs to v_wmma_f32_16x16x32_bf16, f32 accumulate.
// GEMM grids are n-tile-fastest so one m-stripe's A tile stays L2-resident
// across its 24 (qkv) / 8 (proj) consecutive blocks; weights (3 MB) live in L2.

typedef __attribute__((ext_vector_type(16))) __bf16 bf16x16;
typedef __attribute__((ext_vector_type(8)))  __bf16 bf16x8;
typedef __attribute__((ext_vector_type(8)))  float  f32x8;

#define NWB    2048
#define NTOK   49
#define CDIM   512
#define NHEAD  16
#define HDIM   32
#define MTOT   (NWB * NTOK)          // 100352 = 128 * 784
#define REG_ELEMS (51380224ull)      // 2048*16*49*32 == 100352*512

union frag_u { bf16x16 v; bf16x8 h[2]; };

// Load one 16-bit WMMA A/B fragment from an LDS row.
// CDNA5 16-bit layout: lanes 0-15 hold K = 0..7 / 16..23, lanes 16-31 hold
// K = 8..15 / 24..31 (kh = (lane>>4)*8).
__device__ __forceinline__ bf16x16 load_frag(const __bf16* rowbase, int kh) {
    frag_u u;
    u.h[0] = *(const bf16x8*)(rowbase + kh);
    u.h[1] = *(const bf16x8*)(rowbase + kh + 16);
    return u.v;
}

// ---------------------------------------------------------------------------
// Kernel 1: QKV projection.  qkv = x @ qkv_w + qkv_b, scattered to bf16
// head-major Q (pre-scaled), K, V workspaces:  [b][h][n][d].
// Block: 256 threads (8 waves), 128(M) x 64(N) tile, K-steps of 32.
// Grid: (n_tiles=24, m_tiles=784), n fastest for A-tile L2 reuse.
// ---------------------------------------------------------------------------
__global__ __launch_bounds__(256) void qkv_wmma_kernel(
    const float* __restrict__ x,       // [MTOT, 512]
    const float* __restrict__ qkv_w,   // [512, 1536]
    const float* __restrict__ qkv_b,   // [1536]
    __bf16* __restrict__ qws, __bf16* __restrict__ kws, __bf16* __restrict__ vws)
{
    __shared__ __bf16 ldsA[128][32];   // [m][k]
    __shared__ __bf16 ldsB[64][32];    // [n][k]  (W tile transposed)

    const int t    = threadIdx.x;
    const int lane = t & 31;
    const int wave = t >> 5;           // 0..7
    const int n0   = blockIdx.x * 64;
    const int m0   = blockIdx.y * 128;
    const int wm   = (wave >> 1) * 32; // 0,32,64,96
    const int wn   = (wave & 1)  * 32; // 0,32
    const int lm   = lane & 15;
    const int kh   = (lane >> 4) * 8;

    f32x8 acc[2][2];
    for (int j = 0; j < 2; ++j) {
        float bias = qkv_b[n0 + wn + j * 16 + lm];
        for (int i = 0; i < 2; ++i)
            for (int e = 0; e < 8; ++e) acc[i][j][e] = bias;
    }

    for (int k0 = 0; k0 < CDIM; k0 += 32) {
        __syncthreads();
        {   // stage A tile 128x32 (f32 -> bf16): 16 elems/thread
            int r = t >> 1, cb = (t & 1) * 16;
            const float4* src = (const float4*)(x + (size_t)(m0 + r) * CDIM + k0 + cb);
            for (int q = 0; q < 4; ++q) {
                float4 f = src[q];
                ldsA[r][cb + q * 4 + 0] = (__bf16)f.x;
                ldsA[r][cb + q * 4 + 1] = (__bf16)f.y;
                ldsA[r][cb + q * 4 + 2] = (__bf16)f.z;
                ldsA[r][cb + q * 4 + 3] = (__bf16)f.w;
            }
        }
        {   // stage B tile 32(K)x64(N), transposed into [n][k]: 8 elems/thread
            int k = t >> 3, nb = (t & 7) * 8;
            const float4* src = (const float4*)(qkv_w + (size_t)(k0 + k) * 1536 + n0 + nb);
            for (int q = 0; q < 2; ++q) {
                float4 f = src[q];
                ldsB[nb + q * 4 + 0][k] = (__bf16)f.x;
                ldsB[nb + q * 4 + 1][k] = (__bf16)f.y;
                ldsB[nb + q * 4 + 2][k] = (__bf16)f.z;
                ldsB[nb + q * 4 + 3][k] = (__bf16)f.w;
            }
        }
        __syncthreads();

        for (int i = 0; i < 2; ++i) {
            bf16x16 afrag = load_frag(&ldsA[wm + i * 16 + lm][0], kh);
            for (int j = 0; j < 2; ++j) {
                bf16x16 bfrag = load_frag(&ldsB[wn + j * 16 + lm][0], kh);
                acc[i][j] = __builtin_amdgcn_wmma_f32_16x16x32_bf16(
                    false, afrag, false, bfrag, (short)0, acc[i][j], false, false);
            }
        }
    }

    const float qscale = 0.17677669529663687f;   // 1/sqrt(32)
    const int lh = lane >> 4;
    for (int i = 0; i < 2; ++i) {
        for (int j = 0; j < 2; ++j) {
            int gn    = n0 + wn + j * 16 + lm;     // global column in [0,1536)
            int which = gn >> 9;                   // 0:q 1:k 2:v (tile never straddles)
            int hh    = (gn & 511) >> 5;
            int d     = gn & 31;
            __bf16* dst = (which == 0) ? qws : ((which == 1) ? kws : vws);
            float s     = (which == 0) ? qscale : 1.0f;
            for (int e = 0; e < 8; ++e) {
                int m = m0 + wm + i * 16 + e + 8 * lh;
                int b = m / NTOK, n = m - b * NTOK;
                size_t idx = (((size_t)(b * NHEAD + hh) * NTOK + n) * HDIM + d);
                dst[idx] = (__bf16)(acc[i][j][e] * s);
            }
        }
    }
}

// ---------------------------------------------------------------------------
// Kernel 2: attention per (window, head).  One block = 128 threads (4 waves).
// S = Q K^T (WMMA, K=32), +bias +mask, softmax over 49 cols, O = P V (WMMA).
// ---------------------------------------------------------------------------
__global__ __launch_bounds__(128) void attn_wmma_kernel(
    const __bf16* __restrict__ qws, const __bf16* __restrict__ kws,
    const __bf16* __restrict__ vws,
    const float* __restrict__ mask,       // [64,49,49]
    const float* __restrict__ rel_bias,   // [169,16]
    __bf16* __restrict__ ows)             // [MTOT, 512] bf16 (head-concat)
{
    __shared__ __bf16 qb[64][32];
    __shared__ __bf16 kb[64][32];
    __shared__ __bf16 vT[32][64];         // [d][token]
    __shared__ float  sb[64][64];
    __shared__ __bf16 pb[64][64];

    const int t    = threadIdx.x;
    const int lane = t & 31;
    const int wave = t >> 5;
    const int bh   = blockIdx.x;          // b*16 + h
    const int b    = bh >> 4;
    const int h    = bh & 15;
    const int lm   = lane & 15;
    const int kh   = (lane >> 4) * 8;

    const __bf16* qg = qws + (size_t)bh * NTOK * HDIM;
    const __bf16* kg = kws + (size_t)bh * NTOK * HDIM;
    const __bf16* vg = vws + (size_t)bh * NTOK * HDIM;

    {   // stage Q, K (zero-pad rows 49..63), V transposed
        int r = t >> 1, cb = (t & 1) * 16;
        bool valid = r < NTOK;
        for (int e = 0; e < 16; ++e) {
            __bf16 qv = valid ? qg[r * HDIM + cb + e] : (__bf16)0.0f;
            __bf16 kv = valid ? kg[r * HDIM + cb + e] : (__bf16)0.0f;
            __bf16 vv = valid ? vg[r * HDIM + cb + e] : (__bf16)0.0f;
            qb[r][cb + e] = qv;
            kb[r][cb + e] = kv;
            vT[cb + e][r] = vv;
        }
    }
    __syncthreads();

    // S = Q K^T : wave owns 16 rows, 4 col tiles, single K=32 WMMA each.
    f32x8 s[4];
    {
        bf16x16 afrag = load_frag(&qb[wave * 16 + lm][0], kh);
        for (int j = 0; j < 4; ++j) {
            f32x8 z;
            for (int e = 0; e < 8; ++e) z[e] = 0.0f;
            bf16x16 bfrag = load_frag(&kb[j * 16 + lm][0], kh);
            s[j] = __builtin_amdgcn_wmma_f32_16x16x32_bf16(
                false, afrag, false, bfrag, (short)0, z, false, false);
        }
    }

    // + relative-position bias + shift-window mask -> sb
    const int wmask = b & 63;
    const int lh = lane >> 4;
    for (int j = 0; j < 4; ++j) {
        int jj = j * 16 + lm;
        for (int e = 0; e < 8; ++e) {
            int ii = wave * 16 + e + 8 * lh;
            float val;
            if (ii < NTOK && jj < NTOK) {
                int rpi = ((jj / 7) - (ii / 7) + 6) * 13;   // h-coord-only RPI
                val = s[j][e] + rel_bias[rpi * NHEAD + h]
                             + mask[((size_t)wmask * NTOK + ii) * NTOK + jj];
            } else {
                val = -1e30f;
            }
            sb[ii][jj] = val;
        }
    }
    __syncthreads();

    // Row softmax over 49 valid columns; emit bf16 P (zeros in padding).
    if (t < 64) {
        float mx = -1e30f;
        for (int j = 0; j < NTOK; ++j) mx = fmaxf(mx, sb[t][j]);
        float sum = 0.0f;
        for (int j = 0; j < NTOK; ++j) {
            float e = __expf(sb[t][j] - mx);
            sb[t][j] = e;
            sum += e;
        }
        float inv = 1.0f / sum;
        for (int j = 0; j < 64; ++j)
            pb[t][j] = (t < NTOK && j < NTOK) ? (__bf16)(sb[t][j] * inv) : (__bf16)0.0f;
    }
    __syncthreads();

    // O = P V : wave owns 16 rows; 2 col tiles (d), K = 64 in 2 WMMA steps.
    f32x8 o[2];
    for (int j = 0; j < 2; ++j)
        for (int e = 0; e < 8; ++e) o[j][e] = 0.0f;
    for (int ks = 0; ks < 2; ++ks) {
        bf16x16 afrag = load_frag(&pb[wave * 16 + lm][ks * 32], kh);
        for (int j = 0; j < 2; ++j) {
            bf16x16 bfrag = load_frag(&vT[j * 16 + lm][ks * 32], kh);
            o[j] = __builtin_amdgcn_wmma_f32_16x16x32_bf16(
                false, afrag, false, bfrag, (short)0, o[j], false, false);
        }
    }

    for (int j = 0; j < 2; ++j) {
        for (int e = 0; e < 8; ++e) {
            int ii = wave * 16 + e + 8 * lh;
            if (ii < NTOK) {
                size_t idx = ((size_t)b * NTOK + ii) * CDIM + h * HDIM + j * 16 + lm;
                ows[idx] = (__bf16)o[j][e];
            }
        }
    }
}

// ---------------------------------------------------------------------------
// Kernel 3: output projection.  out = attn_out(bf16) @ proj_w + proj_b (f32).
// Block: 256 threads (8 waves), 128x64 tile.  Grid n-fastest (8 x 784).
// ---------------------------------------------------------------------------
__global__ __launch_bounds__(256) void proj_wmma_kernel(
    const __bf16* __restrict__ aows,    // [MTOT, 512] bf16
    const float*  __restrict__ proj_w,  // [512, 512]
    const float*  __restrict__ proj_b,  // [512]
    float* __restrict__ out)            // [MTOT, 512]
{
    __shared__ __bf16 ldsA[128][32];
    __shared__ __bf16 ldsB[64][32];

    const int t    = threadIdx.x;
    const int lane = t & 31;
    const int wave = t >> 5;
    const int n0   = blockIdx.x * 64;
    const int m0   = blockIdx.y * 128;
    const int wm   = (wave >> 1) * 32;
    const int wn   = (wave & 1)  * 32;
    const int lm   = lane & 15;
    const int kh   = (lane >> 4) * 8;

    f32x8 acc[2][2];
    for (int j = 0; j < 2; ++j) {
        float bias = proj_b[n0 + wn + j * 16 + lm];
        for (int i = 0; i < 2; ++i)
            for (int e = 0; e < 8; ++e) acc[i][j][e] = bias;
    }

    for (int k0 = 0; k0 < CDIM; k0 += 32) {
        __syncthreads();
        {   // A tile already bf16: straight 16B copies, 16 elems/thread
            int r = t >> 1, cb = (t & 1) * 16;
            const bf16x8* src = (const bf16x8*)(aows + (size_t)(m0 + r) * CDIM + k0 + cb);
            *(bf16x8*)&ldsA[r][cb]     = src[0];
            *(bf16x8*)&ldsA[r][cb + 8] = src[1];
        }
        {   // B tile f32 -> bf16, transposed to [n][k]: 8 elems/thread
            int k = t >> 3, nb = (t & 7) * 8;
            const float4* src = (const float4*)(proj_w + (size_t)(k0 + k) * CDIM + n0 + nb);
            for (int q = 0; q < 2; ++q) {
                float4 f = src[q];
                ldsB[nb + q * 4 + 0][k] = (__bf16)f.x;
                ldsB[nb + q * 4 + 1][k] = (__bf16)f.y;
                ldsB[nb + q * 4 + 2][k] = (__bf16)f.z;
                ldsB[nb + q * 4 + 3][k] = (__bf16)f.w;
            }
        }
        __syncthreads();

        for (int i = 0; i < 2; ++i) {
            bf16x16 afrag = load_frag(&ldsA[wm + i * 16 + lm][0], kh);
            for (int j = 0; j < 2; ++j) {
                bf16x16 bfrag = load_frag(&ldsB[wn + j * 16 + lm][0], kh);
                acc[i][j] = __builtin_amdgcn_wmma_f32_16x16x32_bf16(
                    false, afrag, false, bfrag, (short)0, acc[i][j], false, false);
            }
        }
    }

    const int lh = lane >> 4;
    for (int i = 0; i < 2; ++i) {
        for (int j = 0; j < 2; ++j) {
            int n = n0 + wn + j * 16 + lm;
            for (int e = 0; e < 8; ++e) {
                int m = m0 + wm + i * 16 + e + 8 * lh;
                out[(size_t)m * CDIM + n] = acc[i][j][e];
            }
        }
    }
}

// ---------------------------------------------------------------------------
extern "C" void kernel_launch(void* const* d_in, const int* in_sizes, int n_in,
                              void* d_out, int out_size, void* d_ws, size_t ws_size,
                              hipStream_t stream) {
    const float* x      = (const float*)d_in[0];
    const float* mask   = (const float*)d_in[1];
    const float* qkv_w  = (const float*)d_in[2];
    const float* qkv_b  = (const float*)d_in[3];
    const float* proj_w = (const float*)d_in[4];
    const float* proj_b = (const float*)d_in[5];
    const float* relb   = (const float*)d_in[6];
    float* out = (float*)d_out;

    // Workspace: 4 equal bf16 regions of 51,380,224 elems (Q, K, V, attn-out).
    char* ws = (char*)d_ws;
    const size_t REG_BYTES = REG_ELEMS * 2ull;
    __bf16* qws = (__bf16*)(ws + 0 * REG_BYTES);
    __bf16* kws = (__bf16*)(ws + 1 * REG_BYTES);
    __bf16* vws = (__bf16*)(ws + 2 * REG_BYTES);
    __bf16* ows = (__bf16*)(ws + 3 * REG_BYTES);

    dim3 gQkv(1536 / 64, MTOT / 128);    // (24, 784), n-tile fastest
    qkv_wmma_kernel<<<gQkv, 256, 0, stream>>>(x, qkv_w, qkv_b, qws, kws, vws);

    attn_wmma_kernel<<<NWB * NHEAD, 128, 0, stream>>>(qws, kws, vws, mask, relb, ows);

    dim3 gProj(CDIM / 64, MTOT / 128);   // (8, 784), n-tile fastest
    proj_wmma_kernel<<<gProj, 256, 0, stream>>>(ows, proj_w, proj_b, out);
}
